// Attention_65025804861926
// MI455X (gfx1250) — matrix operand order repaired and checked by reading, not compiled, not run on hardware
//
#include <hip/hip_runtime.h>
#include <hip/hip_bf16.h>

// ---------------------------------------------------------------------------
// MI455X (gfx1250) bf16-WMMA pipeline for the feature-axis "attention":
//   q = LN(x@wq^T); k = LN(x@wk^T); v = x@wv^T
//   scores[b] = q[b]^T @ k[b]   (contraction over S)
//   out = (softmax(scores) @ v^T)^T @ wo^T
// All heavy math routed through v_wmma_f32_16x16x32_bf16 (fp32 accumulate).
// Round-2 change: software-pipelined global->reg->LDS staging so the
// s_wait_loadcnt lands AFTER the WMMA block (global latency hidden), plus
// global_prefetch for the k+2 tile.
// ---------------------------------------------------------------------------

typedef __bf16 bf16_t;
typedef bf16_t v16bf __attribute__((ext_vector_type(16)));
typedef float  v8f   __attribute__((ext_vector_type(8)));

__device__ __forceinline__ unsigned short f32_to_bf16_rne(float f) {
  unsigned int u = __float_as_uint(f);
  u += 0x7FFFu + ((u >> 16) & 1u);
  return (unsigned short)(u >> 16);
}

// ---------------- fp32 -> bf16 elementwise convert (x4 vectorized) ---------
__global__ __launch_bounds__(256)
void cvt_f32_bf16_kernel(const float* __restrict__ in,
                         unsigned short* __restrict__ out, long long n4) {
  long long i = (long long)blockIdx.x * blockDim.x + threadIdx.x;
  if (i >= n4) return;
  float4 v = ((const float4*)in)[i];
  ushort4 o;
  o.x = f32_to_bf16_rne(v.x);
  o.y = f32_to_bf16_rne(v.y);
  o.z = f32_to_bf16_rne(v.z);
  o.w = f32_to_bf16_rne(v.w);
  ((ushort4*)out)[i] = o;
}

// ---------------- WMMA GEMM: C(m,n) = sum_k A(m,k)*B(k,n) ------------------
// TRA=false: A stored [M][K] (lda=row stride). TRA=true: A stored [K][M].
// TRB=false: B stored [K][N].                  TRB=true: B stored [N][K].
// OUTBF16 selects bf16 vs fp32 output. Batched via blockIdx.z + strides.
template <bool TRA, bool TRB, bool OUTBF16>
__global__ __launch_bounds__(256)
void gemm_wmma_bf16(const unsigned short* __restrict__ A,
                    const unsigned short* __restrict__ B,
                    void* __restrict__ Cv,
                    int M, int N, int K, int lda, int ldb, int ldc,
                    long long strideA, long long strideB, long long strideC) {
  constexpr int BM = 128, BN = 128, BK = 32;
  constexpr int LDK = BK + 8;  // halfword pad: 80B rows keep 16B alignment
  __shared__ unsigned short As[2][BM * LDK];
  __shared__ unsigned short Bs[2][BN * LDK];

  const int batch = blockIdx.z;
  const unsigned short* Ab = A + (long long)batch * strideA;
  const unsigned short* Bb = B + (long long)batch * strideB;

  const int tid   = threadIdx.x;
  const int wave  = tid >> 5;
  const int lane  = tid & 31;
  const int waveM = wave >> 1;  // 0..3 -> 32-row strip
  const int waveN = wave & 1;   // 0..1 -> 64-col strip
  const int blkM  = blockIdx.y * BM;
  const int blkN  = blockIdx.x * BN;
  const int laneN = lane & 15;
  const int hiHalf = lane >> 4;

  v8f acc[2][4];
#pragma unroll
  for (int i = 0; i < 2; ++i)
#pragma unroll
    for (int j = 0; j < 4; ++j)
#pragma unroll
      for (int e = 0; e < 8; ++e) acc[i][j][e] = 0.0f;

  // ---- stage 1: issue global loads into registers (no LDS, no waits) ----
  auto issueA = [&](int k0, uint4& d0, uint4& d1) {
    if constexpr (!TRA) {
      int m  = tid >> 1;
      int ks = (tid & 1) * 16;
      const uint4* src = (const uint4*)(Ab + (long long)(blkM + m) * lda + k0 + ks);
      d0 = src[0]; d1 = src[1];
    } else {
      int k  = tid >> 3;
      int ms = (tid & 7) * 16;
      const uint4* src = (const uint4*)(Ab + (long long)(k0 + k) * lda + blkM + ms);
      d0 = src[0]; d1 = src[1];
    }
  };
  auto issueB = [&](int k0, uint4& d0, uint4& d1) {
    if constexpr (!TRB) {
      int k  = tid >> 3;
      int ns = (tid & 7) * 16;
      const uint4* src = (const uint4*)(Bb + (long long)(k0 + k) * ldb + blkN + ns);
      d0 = src[0]; d1 = src[1];
    } else {
      int n  = tid >> 1;
      int ks = (tid & 1) * 16;
      const uint4* src = (const uint4*)(Bb + (long long)(blkN + n) * ldb + k0 + ks);
      d0 = src[0]; d1 = src[1];
    }
  };

  // ---- stage 3: registers -> LDS (this is where s_wait_loadcnt lands) ----
  auto storeA = [&](int buf, const uint4& d0, const uint4& d1) {
    if constexpr (!TRA) {
      int m  = tid >> 1;
      int ks = (tid & 1) * 16;
      uint4* dst = (uint4*)(&As[buf][m * LDK + ks]);
      dst[0] = d0; dst[1] = d1;
    } else {
      int k  = tid >> 3;
      int ms = (tid & 7) * 16;
      union { uint4 q[2]; unsigned short h[16]; } d;
      d.q[0] = d0; d.q[1] = d1;
#pragma unroll
      for (int i = 0; i < 16; ++i) As[buf][(ms + i) * LDK + k] = d.h[i];
    }
  };
  auto storeB = [&](int buf, const uint4& d0, const uint4& d1) {
    if constexpr (!TRB) {
      int k  = tid >> 3;
      int ns = (tid & 7) * 16;
      union { uint4 q[2]; unsigned short h[16]; } d;
      d.q[0] = d0; d.q[1] = d1;
#pragma unroll
      for (int i = 0; i < 16; ++i) Bs[buf][(ns + i) * LDK + k] = d.h[i];
    } else {
      int n  = tid >> 1;
      int ks = (tid & 1) * 16;
      uint4* dst = (uint4*)(&Bs[buf][n * LDK + ks]);
      dst[0] = d0; dst[1] = d1;
    }
  };

  uint4 ra0, ra1, rb0, rb1;
  issueA(0, ra0, ra1);
  issueB(0, rb0, rb1);
  storeA(0, ra0, ra1);
  storeB(0, rb0, rb1);
  __syncthreads();

  const int nk = K / BK;
  for (int kt = 0; kt < nk; ++kt) {
    const int buf = kt & 1;
    const bool haveNext = (kt + 1 < nk);

    // (1) issue next tile's global loads into regs; WMMAs below hide latency
    if (haveNext) {
      issueA((kt + 1) * BK, ra0, ra1);
      issueB((kt + 1) * BK, rb0, rb1);
    }
    // warm L2 for the k+2 tile (global_prefetch_b8); one line per thread
    if (kt + 2 < nk) {
      const int k2 = (kt + 2) * BK;
      if constexpr (!TRA)
        __builtin_prefetch(Ab + (long long)(blkM + (tid >> 1)) * lda + k2 + (tid & 1) * 16, 0, 0);
      else
        __builtin_prefetch(Ab + (long long)(k2 + (tid >> 3)) * lda + blkM + (tid & 7) * 16, 0, 0);
      if constexpr (!TRB)
        __builtin_prefetch(Bb + (long long)(k2 + (tid >> 3)) * ldb + blkN + (tid & 7) * 16, 0, 0);
      else
        __builtin_prefetch(Bb + (long long)(blkN + (tid >> 1)) * ldb + k2 + (tid & 1) * 16, 0, 0);
    }

    // (2) compute current tile from LDS[buf]
    union Frag { v16bf v; unsigned int u[8]; };
    Frag a[2], bf[4];
#pragma unroll
    for (int mi = 0; mi < 2; ++mi) {
      int m = waveM * 32 + mi * 16 + laneN;
      const unsigned short* row = &As[buf][m * LDK];
#pragma unroll
      for (int v = 0; v < 8; ++v) {
        // CDNA5 16-bit A 16x32 layout: lanes0-15 K={0..7,16..23}, lanes16-31 +8
        int k = (v >> 2) * 16 + hiHalf * 8 + (v & 3) * 2;
        a[mi].u[v] = *(const unsigned int*)(row + k);
      }
    }
#pragma unroll
    for (int ni = 0; ni < 4; ++ni) {
      int n = waveN * 64 + ni * 16 + laneN;
      const unsigned short* row = &Bs[buf][n * LDK];
#pragma unroll
      for (int v = 0; v < 8; ++v) {
        // CDNA5 16-bit B 32x16 layout: lanes0-15 K={0..7,16..23}, lanes16-31 +8
        int k = (v >> 2) * 16 + hiHalf * 8 + (v & 3) * 2;
        bf[ni].u[v] = *(const unsigned int*)(row + k);
      }
    }
#pragma unroll
    for (int mi = 0; mi < 2; ++mi)
#pragma unroll
      for (int ni = 0; ni < 4; ++ni)
        acc[mi][ni] = __builtin_amdgcn_wmma_f32_16x16x32_bf16(
            false, a[mi].v, false, bf[ni].v, (short)0, acc[mi][ni], false, false);

    // (3) drain the global loads into the other LDS buffer. Safe without a
    // pre-barrier: readers of buf^1 finished at the previous __syncthreads.
    if (haveNext) {
      storeA(buf ^ 1, ra0, ra1);
      storeB(buf ^ 1, rb0, rb1);
    }
    __syncthreads();
  }

  const long long cOff = (long long)batch * strideC;
#pragma unroll
  for (int mi = 0; mi < 2; ++mi) {
#pragma unroll
    for (int ni = 0; ni < 4; ++ni) {
      int nCol = blkN + waveN * 64 + ni * 16 + laneN;
#pragma unroll
      for (int r = 0; r < 8; ++r) {
        int mRow = blkM + waveM * 32 + mi * 16 + hiHalf * 8 + r;
        float val = acc[mi][ni][r];
        if constexpr (OUTBF16) {
          ((unsigned short*)Cv)[cOff + (long long)mRow * ldc + nCol] = f32_to_bf16_rne(val);
        } else {
          ((float*)Cv)[cOff + (long long)mRow * ldc + nCol] = val;
        }
      }
    }
  }
}

// ---------------- Row LayerNorm (fp32 in -> bf16 out), D cols --------------
__global__ __launch_bounds__(256)
void layernorm_bf16_kernel(const float* __restrict__ X,
                           const float* __restrict__ gamma,
                           const float* __restrict__ beta,
                           unsigned short* __restrict__ Y, int D) {
  const long long row = blockIdx.x;
  const float* x = X + row * D;
  unsigned short* y = Y + row * D;
  float s = 0.f, ss = 0.f;
  for (int i = threadIdx.x; i < D; i += 256) { float v = x[i]; s += v; ss += v * v; }
#pragma unroll
  for (int off = 16; off > 0; off >>= 1) {
    s  += __shfl_xor(s, off, 32);
    ss += __shfl_xor(ss, off, 32);
  }
  __shared__ float rs[8], rss[8];
  int w = threadIdx.x >> 5, ln = threadIdx.x & 31;
  if (ln == 0) { rs[w] = s; rss[w] = ss; }
  __syncthreads();
  if (threadIdx.x == 0) {
    float a = 0.f, b = 0.f;
    for (int i = 0; i < 8; ++i) { a += rs[i]; b += rss[i]; }
    rs[0] = a; rss[0] = b;
  }
  __syncthreads();
  const float mean = rs[0] / (float)D;
  const float var  = rss[0] / (float)D - mean * mean;
  const float inv  = rsqrtf(var + 1e-5f);
  for (int i = threadIdx.x; i < D; i += 256) {
    float v = (x[i] - mean) * inv * gamma[i] + beta[i];
    y[i] = f32_to_bf16_rne(v);
  }
}

// ---------------- Row softmax (fp32 in -> bf16 out), D cols ----------------
__global__ __launch_bounds__(256)
void softmax_bf16_kernel(const float* __restrict__ X,
                         unsigned short* __restrict__ Y, int D) {
  const long long row = blockIdx.x;
  const float* x = X + row * D;
  unsigned short* y = Y + row * D;
  float m = -3.4e38f;
  for (int i = threadIdx.x; i < D; i += 256) m = fmaxf(m, x[i]);
#pragma unroll
  for (int off = 16; off > 0; off >>= 1) m = fmaxf(m, __shfl_xor(m, off, 32));
  __shared__ float rm[8], rsum[8];
  int w = threadIdx.x >> 5, ln = threadIdx.x & 31;
  if (ln == 0) rm[w] = m;
  __syncthreads();
  if (threadIdx.x == 0) {
    float a = rm[0];
    for (int i = 1; i < 8; ++i) a = fmaxf(a, rm[i]);
    rm[0] = a;
  }
  __syncthreads();
  m = rm[0];
  float s = 0.f;
  for (int i = threadIdx.x; i < D; i += 256) s += __expf(x[i] - m);
#pragma unroll
  for (int off = 16; off > 0; off >>= 1) s += __shfl_xor(s, off, 32);
  if (ln == 0) rsum[w] = s;
  __syncthreads();
  if (threadIdx.x == 0) {
    float a = 0.f;
    for (int i = 0; i < 8; ++i) a += rsum[i];
    rsum[0] = a;
  }
  __syncthreads();
  const float invs = 1.0f / rsum[0];
  for (int i = threadIdx.x; i < D; i += 256)
    y[i] = f32_to_bf16_rne(__expf(x[i] - m) * invs);
}

// ---------------------------------------------------------------------------
extern "C" void kernel_launch(void* const* d_in, const int* in_sizes, int n_in,
                              void* d_out, int out_size, void* d_ws, size_t ws_size,
                              hipStream_t stream) {
  (void)in_sizes; (void)n_in; (void)out_size; (void)ws_size;
  constexpr int B = 4, S = 8192, D = 1024;
  constexpr long long SD = (long long)S * D;       // 8,388,608
  constexpr long long BSD = (long long)B * SD;     // 33,554,432
  constexpr long long DD = (long long)D * D;       // 1,048,576

  const float* x       = (const float*)d_in[0];
  const float* wq      = (const float*)d_in[1];
  const float* wk      = (const float*)d_in[2];
  const float* wv      = (const float*)d_in[3];
  const float* wo      = (const float*)d_in[4];
  const float* q_gamma = (const float*)d_in[5];
  const float* q_beta  = (const float*)d_in[6];
  const float* k_gamma = (const float*)d_in[7];
  const float* k_beta  = (const float*)d_in[8];
  float* out = (float*)d_out;

  // Workspace carve-up (bytes)
  char* ws = (char*)d_ws;
  unsigned short* xb   = (unsigned short*)(ws);                       // 64 MB
  unsigned short* wqb  = (unsigned short*)(ws + (BSD * 2));           // 2 MB
  unsigned short* wkb  = (unsigned short*)(ws + (BSD * 2 + DD * 2));
  unsigned short* wvb  = (unsigned short*)(ws + (BSD * 2 + DD * 4));
  unsigned short* wob  = (unsigned short*)(ws + (BSD * 2 + DD * 6));
  float*          tmpf = (float*)(ws + (BSD * 2 + DD * 8));           // 128 MB
  char* p = ws + (BSD * 2 + DD * 8 + BSD * 4);
  unsigned short* qln  = (unsigned short*)(p);            p += BSD * 2;
  unsigned short* kln  = (unsigned short*)(p);            p += BSD * 2;
  unsigned short* vb   = (unsigned short*)(p);            p += BSD * 2;
  float*          scf  = (float*)(p);                     p += (long long)B * DD * 4;
  unsigned short* attn = (unsigned short*)(p);            p += (long long)B * DD * 2;
  unsigned short* ctx  = (unsigned short*)(p);

  const dim3 blk(256);

  // 1) fp32 -> bf16 converts
  cvt_f32_bf16_kernel<<<dim3((unsigned)(BSD / 4 / 256)), blk, 0, stream>>>(x, xb, BSD / 4);
  cvt_f32_bf16_kernel<<<dim3((unsigned)(DD / 4 / 256)), blk, 0, stream>>>(wq, wqb, DD / 4);
  cvt_f32_bf16_kernel<<<dim3((unsigned)(DD / 4 / 256)), blk, 0, stream>>>(wk, wkb, DD / 4);
  cvt_f32_bf16_kernel<<<dim3((unsigned)(DD / 4 / 256)), blk, 0, stream>>>(wv, wvb, DD / 4);
  cvt_f32_bf16_kernel<<<dim3((unsigned)(DD / 4 / 256)), blk, 0, stream>>>(wo, wob, DD / 4);

  const int M1 = B * S;  // 32768 rows for the projection GEMMs
  const dim3 gProj(D / 128, M1 / 128, 1);

  // 2) q = LN(x @ wq^T)
  gemm_wmma_bf16<false, true, false><<<gProj, blk, 0, stream>>>(
      xb, wqb, tmpf, M1, D, D, D, D, D, 0, 0, 0);
  layernorm_bf16_kernel<<<dim3(M1), blk, 0, stream>>>(tmpf, q_gamma, q_beta, qln, D);

  // 3) k = LN(x @ wk^T)   (reuse fp32 scratch)
  gemm_wmma_bf16<false, true, false><<<gProj, blk, 0, stream>>>(
      xb, wkb, tmpf, M1, D, D, D, D, D, 0, 0, 0);
  layernorm_bf16_kernel<<<dim3(M1), blk, 0, stream>>>(tmpf, k_gamma, k_beta, kln, D);

  // 4) v = x @ wv^T  (bf16 out, feeds next GEMM)
  gemm_wmma_bf16<false, true, true><<<gProj, blk, 0, stream>>>(
      xb, wvb, vb, M1, D, D, D, D, D, 0, 0, 0);

  // 5) scores[b] = q[b]^T @ k[b]  -> [B,D,D] fp32 ; K = S = 8192
  gemm_wmma_bf16<true, false, false><<<dim3(D / 128, D / 128, B), blk, 0, stream>>>(
      qln, kln, scf, D, D, S, D, D, D, SD, SD, DD);

  // 6) attn = softmax(scores) -> bf16
  softmax_bf16_kernel<<<dim3(B * D), blk, 0, stream>>>(scf, attn, D);

  // 7) ctx[b] = v[b] @ attn[b]^T   (out[s,i] = sum_j attn[i,j]*v[s,j])
  gemm_wmma_bf16<false, true, true><<<dim3(D / 128, S / 128, B), blk, 0, stream>>>(
      vb, attn, ctx, S, D, D, D, D, D, SD, DD, SD);

  // 8) final = ctx @ wo^T -> fp32 output
  gemm_wmma_bf16<false, true, false><<<gProj, blk, 0, stream>>>(
      ctx, wob, out, M1, D, D, D, D, D, 0, 0, 0);
}
